// DecisionVQVAE_1116691497625
// MI455X (gfx1250) — compile-verified, matrix operand order checked by reading, not run on hardware
//
#include <hip/hip_runtime.h>

// ---------------------------------------------------------------------------
// Types for CDNA5 WMMA (wave32)
// ---------------------------------------------------------------------------
typedef __attribute__((ext_vector_type(16))) __bf16 v16bf;
typedef __attribute__((ext_vector_type(8)))  __bf16 v8bf;
typedef __attribute__((ext_vector_type(4)))  __bf16 v4bf;
typedef __attribute__((ext_vector_type(8)))  float  v8f;

union FragBF {
    v16bf v;
    v8bf  h[2];
};

__device__ inline v4bf cvt4(float4 a) {
    v4bf r;
    r.x = (__bf16)a.x; r.y = (__bf16)a.y; r.z = (__bf16)a.z; r.w = (__bf16)a.w;
    return r;
}
__device__ inline v8bf cvt8(float4 a, float4 b) {
    v8bf r;
    r[0] = (__bf16)a.x; r[1] = (__bf16)a.y; r[2] = (__bf16)a.z; r[3] = (__bf16)a.w;
    r[4] = (__bf16)b.x; r[5] = (__bf16)b.y; r[6] = (__bf16)b.z; r[7] = (__bf16)b.w;
    return r;
}

__device__ inline v8f wmma_bf16(const FragBF& a, const FragBF& b, v8f c) {
    // D = A(16x32 bf16) * B(32x16 bf16) + C(16x16 f32)
    return __builtin_amdgcn_wmma_f32_16x16x32_bf16(
        /*neg_a=*/false, a.v, /*neg_b=*/false, b.v,
        /*c_mod=*/(short)0, c, /*reuse_a=*/false, /*reuse_b=*/false);
}

// Async global->LDS 16B copy (CDNA5, tracked by ASYNCcnt).
// ldsOff: LDS byte offset (low 32 bits of the flat LDS address),
// gaddr : 64-bit global address.
__device__ inline void async_copy_b128(unsigned ldsOff, const void* gaddr) {
    asm volatile("global_load_async_to_lds_b128 %0, %1, off"
                 :: "v"(ldsOff), "v"((unsigned long long)(uintptr_t)gaddr)
                 : "memory");
}
__device__ inline void async_wait0() {
    asm volatile("s_wait_asynccnt 0x0" ::: "memory");
}

// ---------------------------------------------------------------------------
// Problem constants
// ---------------------------------------------------------------------------
#define NROWS  131072
#define D_IN   768
#define D_HID  512
#define D_CODE 256
#define KCODES 512

// GEMM tiling
#define BM 128
#define BN 128
#define BK 32
#define ASTR 40   // padded LDS strides (elements) -> 80B rows, 16B-aligned frags
#define BSTR 40

// ---------------------------------------------------------------------------
// Tiled WMMA GEMM:  out[M,N] = act( A[M,K] * B[K,N] + bias[N] )
//   A: f32 (converted to bf16 during LDS staging) or bf16 (async DMA to LDS)
//   B: f32 row-major (converted during staging)
//   out: f32 or bf16
// Block = 256 threads (8 waves), workgroup tile 128x128, wave tile 32x64.
// ---------------------------------------------------------------------------
template<bool A_BF16, bool RELU, bool OUT_BF16>
__global__ __launch_bounds__(256)
void gemm_wmma(const void* __restrict__ Ap, const float* __restrict__ B,
               const float* __restrict__ bias, void* __restrict__ outp,
               int K, int N)
{
    __shared__ __bf16 ldsA[BM * ASTR];
    __shared__ __bf16 ldsB[BN * BSTR];

    const int tid  = threadIdx.x;
    const int wave = tid >> 5;
    const int lane = tid & 31;
    const int hi   = lane >> 4;   // half-wave select
    const int lr   = lane & 15;
    const int waveM = wave & 3;   // 4 M-groups of 32 rows
    const int waveN = wave >> 2;  // 2 N-groups of 64 cols
    const int rowBase = blockIdx.y * BM;
    const int colBase = blockIdx.x * BN;

    v8f acc[2][4];
    const v8f vzero = {0.f, 0.f, 0.f, 0.f, 0.f, 0.f, 0.f, 0.f};
#pragma unroll
    for (int mt = 0; mt < 2; ++mt)
#pragma unroll
        for (int nt = 0; nt < 4; ++nt)
            acc[mt][nt] = vzero;

    const int ksteps = K / BK;
    for (int kt = 0; kt < ksteps; ++kt) {
        // ---- stage A tile (BM x BK) into LDS as bf16 -----------------------
        if (A_BF16) {
            // raw bf16 copy: use CDNA5 async global->LDS DMA (ASYNCcnt)
            const __bf16* A = (const __bf16*)Ap;
#pragma unroll
            for (int i = 0; i < 2; ++i) {
                int lin = tid + i * 256;          // 512 chunks of 8 bf16
                int row = lin >> 2, cv = lin & 3;
                const __bf16* gp = A + (size_t)(rowBase + row) * K + kt * BK + cv * 8;
                unsigned ldsOff = (unsigned)(uintptr_t)(&ldsA[row * ASTR + cv * 8]);
                async_copy_b128(ldsOff, gp);
            }
        } else {
            const float* A = (const float*)Ap;
#pragma unroll
            for (int i = 0; i < 4; ++i) {
                int lin = tid + i * 256;          // 1024 float4 chunks
                int row = lin >> 3, cv = lin & 7;
                float4 v = *(const float4*)(A + (size_t)(rowBase + row) * K + kt * BK + cv * 4);
                *(v4bf*)(&ldsA[row * ASTR + cv * 4]) = cvt4(v);
            }
        }
        // ---- stage B tile (BK x BN) into LDS transposed: [n][k] ------------
        // (overlaps with the in-flight async A copy in the A_BF16 case)
#pragma unroll
        for (int i = 0; i < 4; ++i) {
            int lin = tid + i * 256;              // 1024 (4-k x 1-n) groups
            int kg = lin >> 7, n = lin & 127;
            int k = kg * 4;
            const float* bp = B + (size_t)(kt * BK + k) * N + colBase + n;
            float4 f;
            f.x = bp[0];
            f.y = bp[(size_t)N];
            f.z = bp[(size_t)2 * N];
            f.w = bp[(size_t)3 * N];
            *(v4bf*)(&ldsB[n * BSTR + k]) = cvt4(f);
        }
        if (A_BF16) async_wait0();                // drain ASYNCcnt before barrier
        __syncthreads();

        // ---- load fragments ------------------------------------------------
        FragBF aF[2], bF[4];
        const int c0 = hi * 8;
#pragma unroll
        for (int mt = 0; mt < 2; ++mt) {
            int m = waveM * 32 + mt * 16 + lr;
            aF[mt].h[0] = *(const v8bf*)(&ldsA[m * ASTR + c0]);
            aF[mt].h[1] = *(const v8bf*)(&ldsA[m * ASTR + c0 + 16]);
        }
#pragma unroll
        for (int nt = 0; nt < 4; ++nt) {
            int n = waveN * 64 + nt * 16 + lr;
            bF[nt].h[0] = *(const v8bf*)(&ldsB[n * BSTR + c0]);
            bF[nt].h[1] = *(const v8bf*)(&ldsB[n * BSTR + c0 + 16]);
        }
        // ---- 8 WMMAs -------------------------------------------------------
#pragma unroll
        for (int mt = 0; mt < 2; ++mt)
#pragma unroll
            for (int nt = 0; nt < 4; ++nt)
                acc[mt][nt] = wmma_bf16(aF[mt], bF[nt], acc[mt][nt]);
        __syncthreads();
    }

    // ---- epilogue: bias (+relu), store -------------------------------------
#pragma unroll
    for (int mt = 0; mt < 2; ++mt) {
#pragma unroll
        for (int nt = 0; nt < 4; ++nt) {
            int col  = colBase + waveN * 64 + nt * 16 + lr;
            int row0 = rowBase + waveM * 32 + mt * 16 + hi * 8;
            float bi = bias[col];
#pragma unroll
            for (int r = 0; r < 8; ++r) {
                float v = acc[mt][nt][r] + bi;
                if (RELU) v = fmaxf(v, 0.0f);
                if (OUT_BF16)
                    ((__bf16*)outp)[(size_t)(row0 + r) * N + col] = (__bf16)v;
                else
                    ((float*)outp)[(size_t)(row0 + r) * N + col] = v;
            }
        }
    }
}

// ---------------------------------------------------------------------------
// Codebook squared norms: cnorm[k] = sum_d codebook[k][d]^2
// ---------------------------------------------------------------------------
__global__ __launch_bounds__(256)
void cnorm_kernel(const float* __restrict__ cb, float* __restrict__ cnorm)
{
    int g = blockIdx.x * 256 + threadIdx.x;
    if (g < KCODES) {
        const float* r = cb + (size_t)g * D_CODE;
        float s = 0.f;
#pragma unroll 4
        for (int i = 0; i < D_CODE; i += 4) {
            float4 v = *(const float4*)(r + i);
            s += v.x * v.x + v.y * v.y + v.z * v.z + v.w * v.w;
        }
        cnorm[g] = s;
    }
}

// ---------------------------------------------------------------------------
// Fused: dots = z @ codebook^T via WMMA, dist = ||c||^2 - 2*dots, argmin,
// gather quantized -> bf16, commitment-loss partial per wave (deterministic).
// Block = 256 threads (8 waves); each wave owns 16 rows; block owns 128 rows.
// ---------------------------------------------------------------------------
#define CSTR 264   // padded LDS stride for a 16-code tile (16B-aligned chunks)

__global__ __launch_bounds__(256)
void vq_argmin_kernel(const float* __restrict__ z, const float* __restrict__ cb,
                      const float* __restrict__ cnorm, int* __restrict__ outIdx,
                      void* __restrict__ qp, float* __restrict__ partials)
{
    __shared__ __bf16 ldsC[16 * CSTR];
    __bf16* q = (__bf16*)qp;

    const int tid  = threadIdx.x;
    const int wave = tid >> 5;
    const int lane = tid & 31;
    const int hi   = lane >> 4;
    const int lr   = lane & 15;
    const int rowBase = blockIdx.x * 128 + wave * 16;

    // ---- load z A-fragments (16 rows x 256 K) into registers, once ---------
    FragBF aF[8];
    {
        const float* zr = z + (size_t)(rowBase + lr) * D_CODE;
#pragma unroll
        for (int ks = 0; ks < 8; ++ks) {
            int c0 = ks * 32 + hi * 8;
            float4 a0 = *(const float4*)(zr + c0);
            float4 a1 = *(const float4*)(zr + c0 + 4);
            float4 b0 = *(const float4*)(zr + c0 + 16);
            float4 b1 = *(const float4*)(zr + c0 + 20);
            aF[ks].h[0] = cvt8(a0, a1);
            aF[ks].h[1] = cvt8(b0, b1);
        }
    }

    float minv[8];
    int   mini[8];
#pragma unroll
    for (int r = 0; r < 8; ++r) { minv[r] = 3.402823466e38f; mini[r] = 0; }

    const v8f vzero = {0.f, 0.f, 0.f, 0.f, 0.f, 0.f, 0.f, 0.f};

    for (int t = 0; t < KCODES / 16; ++t) {
        // ---- stage 16-code tile of codebook into LDS as bf16 ---------------
#pragma unroll
        for (int i = 0; i < 4; ++i) {
            int lin = tid + i * 256;              // 1024 float4 chunks
            int code = lin >> 6, cv = lin & 63;
            float4 v = *(const float4*)(cb + (size_t)(t * 16 + code) * D_CODE + cv * 4);
            *(v4bf*)(&ldsC[code * CSTR + cv * 4]) = cvt4(v);
        }
        __syncthreads();

        v8f acc = vzero;
#pragma unroll
        for (int ks = 0; ks < 8; ++ks) {
            FragBF bF;
            int c0 = ks * 32 + hi * 8;
            bF.h[0] = *(const v8bf*)(&ldsC[lr * CSTR + c0]);
            bF.h[1] = *(const v8bf*)(&ldsC[lr * CSTR + c0 + 16]);
            acc = wmma_bf16(aF[ks], bF, acc);
        }

        int   code = t * 16 + lr;
        float cn   = cnorm[code];
#pragma unroll
        for (int r = 0; r < 8; ++r) {
            float d = cn - 2.0f * acc[r];
            if (d < minv[r]) { minv[r] = d; mini[r] = code; }  // ties -> first index
        }
        __syncthreads();
    }

    // ---- cross-lane argmin within each 16-lane half ------------------------
#pragma unroll
    for (int r = 0; r < 8; ++r) {
#pragma unroll
        for (int off = 1; off < 16; off <<= 1) {
            float ov = __shfl_xor(minv[r], off, 32);
            int   oi = __shfl_xor(mini[r], off, 32);
            if (ov < minv[r] || (ov == minv[r] && oi < mini[r])) {
                minv[r] = ov; mini[r] = oi;
            }
        }
    }
    if (lr == 0) {
#pragma unroll
        for (int r = 0; r < 8; ++r)
            outIdx[rowBase + hi * 8 + r] = mini[r];
    }

    // ---- gather quantized rows (bf16) + commitment-loss partial ------------
    float lsum = 0.f;
#pragma unroll
    for (int rr = 0; rr < 16; ++rr) {
        int idx = __shfl(mini[rr & 7], (rr < 8) ? 0 : 16, 32);
        int row = rowBase + rr;
        const float* cp = cb + (size_t)idx * D_CODE + lane * 8;
        const float* zp = z  + (size_t)row * D_CODE + lane * 8;
        float4 c0 = *(const float4*)(cp);
        float4 c1 = *(const float4*)(cp + 4);
        float4 z0 = *(const float4*)(zp);
        float4 z1 = *(const float4*)(zp + 4);
        float d;
        d = c0.x - z0.x; lsum += d * d;
        d = c0.y - z0.y; lsum += d * d;
        d = c0.z - z0.z; lsum += d * d;
        d = c0.w - z0.w; lsum += d * d;
        d = c1.x - z1.x; lsum += d * d;
        d = c1.y - z1.y; lsum += d * d;
        d = c1.z - z1.z; lsum += d * d;
        d = c1.w - z1.w; lsum += d * d;
        *(v8bf*)(q + (size_t)row * D_CODE + lane * 8) = cvt8(c0, c1);
    }
#pragma unroll
    for (int off = 1; off < 32; off <<= 1)
        lsum += __shfl_xor(lsum, off, 32);
    if (lane == 0)
        partials[blockIdx.x * 8 + wave] = lsum;
}

// ---------------------------------------------------------------------------
// Deterministic commit-loss finalize: sum 8192 per-wave partials / (N*D_CODE)
// ---------------------------------------------------------------------------
__global__ __launch_bounds__(256)
void finalize_kernel(const float* __restrict__ partials, float* __restrict__ outLoss)
{
    __shared__ float red[256];
    float s = 0.f;
    for (int i = threadIdx.x; i < 8192; i += 256) s += partials[i];
    red[threadIdx.x] = s;
    __syncthreads();
    for (int st = 128; st > 0; st >>= 1) {
        if (threadIdx.x < st) red[threadIdx.x] += red[threadIdx.x + st];
        __syncthreads();
    }
    if (threadIdx.x == 0)
        *outLoss = red[0] / ((float)NROWS * (float)D_CODE);
}

// ---------------------------------------------------------------------------
// Launch
// ---------------------------------------------------------------------------
extern "C" void kernel_launch(void* const* d_in, const int* in_sizes, int n_in,
                              void* d_out, int out_size, void* d_ws, size_t ws_size,
                              hipStream_t stream)
{
    (void)in_sizes; (void)n_in; (void)out_size; (void)ws_size;

    const float* x  = (const float*)d_in[0];
    const float* W1 = (const float*)d_in[1];
    const float* b1 = (const float*)d_in[2];
    const float* W2 = (const float*)d_in[3];
    const float* b2 = (const float*)d_in[4];
    const float* cb = (const float*)d_in[5];
    const float* W3 = (const float*)d_in[6];
    const float* b3 = (const float*)d_in[7];
    const float* W4 = (const float*)d_in[8];
    const float* b4 = (const float*)d_in[9];

    // workspace layout
    char* ws = (char*)d_ws;
    const size_t H_BYTES = (size_t)NROWS * D_HID * 2;   // bf16 hidden (reused h1/h2)
    const size_t Z_BYTES = (size_t)NROWS * D_CODE * 4;  // f32 z
    const size_t Q_BYTES = (size_t)NROWS * D_CODE * 2;  // bf16 quantized
    void*  h        = (void*)(ws);
    float* z        = (float*)(ws + H_BYTES);
    void*  q        = (void*)(ws + H_BYTES + Z_BYTES);
    float* cnrm     = (float*)(ws + H_BYTES + Z_BYTES + Q_BYTES);
    float* partials = (float*)(ws + H_BYTES + Z_BYTES + Q_BYTES + 4096);

    // outputs: recon [N,768] f32, indices [N] i32, loss [1] f32
    float* recon   = (float*)d_out;
    int*   outIdx  = (int*)((float*)d_out + (size_t)NROWS * D_IN);
    float* outLoss = (float*)d_out + (size_t)NROWS * D_IN + NROWS;

    // 1) codebook norms
    cnorm_kernel<<<2, 256, 0, stream>>>(cb, cnrm);
    // 2) h1 = relu(x @ W1 + b1)            [N,512] bf16
    gemm_wmma<false, true, true><<<dim3(D_HID / BN, NROWS / BM), 256, 0, stream>>>(
        x, W1, b1, h, D_IN, D_HID);
    // 3) z = h1 @ W2 + b2                  [N,256] f32
    gemm_wmma<true, false, false><<<dim3(D_CODE / BN, NROWS / BM), 256, 0, stream>>>(
        h, W2, b2, z, D_HID, D_CODE);
    // 4) argmin over codebook + quantize + commit-loss partials
    vq_argmin_kernel<<<NROWS / 128, 256, 0, stream>>>(z, cb, cnrm, outIdx, q, partials);
    // 5) h2 = relu(q @ W3 + b3)            [N,512] bf16 (reuses h buffer)
    gemm_wmma<true, true, true><<<dim3(D_HID / BN, NROWS / BM), 256, 0, stream>>>(
        q, W3, b3, h, D_CODE, D_HID);
    // 6) recon = h2 @ W4 + b4              [N,768] f32 -> d_out
    gemm_wmma<true, false, false><<<dim3(D_IN / BN, NROWS / BM), 256, 0, stream>>>(
        h, W4, b4, recon, D_HID, D_IN);
    // 7) loss finalize (deterministic tree reduce)
    finalize_kernel<<<1, 256, 0, stream>>>(partials, outLoss);
}
